// CrossAttention_83442624626717
// MI455X (gfx1250) — matrix operand order repaired and checked by reading, not compiled, not run on hardware
//
#include <hip/hip_runtime.h>
#include <stdint.h>

// ---------------------------------------------------------------------------
// CDNA5 (gfx1250) wave32 WMMA types
// ---------------------------------------------------------------------------
typedef __attribute__((ext_vector_type(16))) __bf16    v16bf;
typedef __attribute__((ext_vector_type(8)))  float     v8f;
typedef __attribute__((ext_vector_type(4)))  uint32_t  u32x4;
typedef __attribute__((ext_vector_type(2)))  float     f32x2;

union Frag {              // 8 VGPRs = 16 bf16 (A or B operand of 16x16x32 bf16 WMMA)
  v16bf v;
  u32x4 q[2];
};

__device__ __forceinline__ uint16_t f32_to_bf16(float f) {
  uint32_t u = __float_as_uint(f);
  return (uint16_t)((u + 0x7FFFu + ((u >> 16) & 1u)) >> 16);  // RNE
}
__device__ __forceinline__ uint32_t pack_bf16x2(float lo, float hi) {
  return (uint32_t)f32_to_bf16(lo) | ((uint32_t)f32_to_bf16(hi) << 16);
}

#define B_   2
#define H_   8
#define T_   2048
#define HD_  64
#define C_   512
#define MTOT 4096      // B*T rows for all projection GEMMs

// ---------------------------------------------------------------------------
// Projection GEMM: Y = X(f32)[M,512] @ W(f32)[512,512] + bias, output bf16.
// layout 0: head-major  [B,H,T,64]   (Q and K)
// layout 1: transposed  [B,H,64,T]   (V — so P@V B-fragments are contiguous)
// ---------------------------------------------------------------------------
__global__ __launch_bounds__(256)
void proj_gemm_kernel(const float* __restrict__ X, const float* __restrict__ W,
                      const float* __restrict__ bias, uint16_t* __restrict__ out,
                      int layout) {
  __shared__ __align__(16) uint16_t Xs[64][40];   // [M][K] bf16, 80B rows
  __shared__ __align__(16) uint16_t Ws[64][40];   // [N][K] bf16 (transposed)

  const int tid  = threadIdx.x;
  const int wave = tid >> 5, lane = tid & 31, hi = lane >> 4, ln = lane & 15;
  const int n0 = blockIdx.x * 64, m0 = blockIdx.y * 64;
  const int mt  = wave & 3;
  const int ntb = (wave >> 2) * 2;

  v8f acc[2];
#pragma unroll
  for (int j = 0; j < 2; ++j)
#pragma unroll
    for (int i = 0; i < 8; ++i) acc[j][i] = 0.f;

  for (int kt = 0; kt < C_; kt += 32) {
#pragma unroll
    for (int i = 0; i < 4; ++i) {                  // 1024 dword slots each array
      int idx = tid + i * 256;
      // Xs: pack two consecutive k into one dword store.
      int m = idx >> 4, kd = idx & 15;
      f32x2 xv = *(const f32x2*)(X + (size_t)(m0 + m) * C_ + kt + kd * 2);
      ((uint32_t*)&Xs[m][0])[kd] = pack_bf16x2(xv.x, xv.y);
      // Ws (transposed): two coalesced row reads, packed dword store.
      int n = idx & 63, kw = idx >> 6;
      float w0 = W[(size_t)(kt + kw * 2) * C_ + n0 + n];
      float w1 = W[(size_t)(kt + kw * 2 + 1) * C_ + n0 + n];
      ((uint32_t*)&Ws[n][0])[kw] = pack_bf16x2(w0, w1);
    }
    __syncthreads();

    Frag a, b0, b1;
    const uint32_t* ar = (const uint32_t*)&Xs[mt * 16 + ln][0];
    a.q[0] = *(const u32x4*)(ar + hi * 4);
    a.q[1] = *(const u32x4*)(ar + 8 + hi * 4);
    const uint32_t* br0 = (const uint32_t*)&Ws[ntb * 16 + ln][0];
    b0.q[0] = *(const u32x4*)(br0 + hi * 4);
    b0.q[1] = *(const u32x4*)(br0 + 8 + hi * 4);
    const uint32_t* br1 = (const uint32_t*)&Ws[(ntb + 1) * 16 + ln][0];
    b1.q[0] = *(const u32x4*)(br1 + hi * 4);
    b1.q[1] = *(const u32x4*)(br1 + 8 + hi * 4);

    acc[0] = __builtin_amdgcn_wmma_f32_16x16x32_bf16(false, a.v, false, b0.v,
                                                     (short)0, acc[0], false, false);
    acc[1] = __builtin_amdgcn_wmma_f32_16x16x32_bf16(false, a.v, false, b1.v,
                                                     (short)0, acc[1], false, false);
    __syncthreads();
  }

#pragma unroll
  for (int j = 0; j < 2; ++j) {
    int gn = n0 + (ntb + j) * 16 + ln;
    float bvv = bias[gn];
    int h = gn >> 6, d = gn & 63;
#pragma unroll
    for (int r = 0; r < 8; ++r) {
      int gm = m0 + mt * 16 + hi * 8 + r;
      int bb = gm >> 11, t = gm & (T_ - 1);
      float val = acc[j][r] + bvv;
      size_t o = (layout == 0)
               ? ((size_t)((bb * H_ + h) * T_ + t) * HD_ + d)
               : ((size_t)((bb * H_ + h) * HD_ + d) * T_ + t);
      out[o] = f32_to_bf16(val);
    }
  }
}

// ---------------------------------------------------------------------------
// Flash attention. Block = 8 waves, all on the SAME (b,h), consecutive q-tiles.
// Each 32-key step: K-tile (32x64) and V-tile (64x32, d-major) staged into LDS
// once per block with GLOBAL_LOAD_ASYNC_TO_LDS_B128 (ASYNCcnt path, no VGPR
// round-trip), then all 8 waves read WMMA fragments from LDS. 8x less global
// read traffic than per-wave streaming.
// ---------------------------------------------------------------------------
__global__ __launch_bounds__(256)
void attn_kernel(const uint16_t* __restrict__ qbuf, const uint16_t* __restrict__ kbuf,
                 const uint16_t* __restrict__ vbuf, const int* __restrict__ pmask,
                 uint16_t* __restrict__ abuf) {
  __shared__ __align__(16) uint16_t Ks[32][72];     // [s_local][d], 144B rows
  __shared__ __align__(16) uint16_t Vs[64][40];     // [d][s_local], 80B rows
  __shared__ __align__(16) uint16_t Ps[8][16][40];  // per-wave P tile

  const int tid  = threadIdx.x;
  const int wave = tid >> 5, lane = tid & 31, hi = lane >> 4, ln = lane & 15;
  const int blk = blockIdx.x;                 // 256 blocks
  const int b   = blk >> 7;
  const int rem = blk & 127;
  const int h   = rem >> 4;
  const int q0  = ((rem & 15) << 7) + wave * 16;   // this wave's 16 query rows

  const uint16_t* kbase = kbuf + (size_t)(b * H_ + h) * T_ * HD_;
  const uint16_t* vbase = vbuf + (size_t)(b * H_ + h) * HD_ * T_;
  const int* pm = pmask + (size_t)b * T_;

  // Q A-fragments (d-chunks {0..31},{32..63}); resident in VGPRs all kernel.
  const uint32_t* qr =
      (const uint32_t*)(qbuf + ((size_t)(b * H_ + h) * T_ + q0 + ln) * HD_);
  Frag qf[2];
#pragma unroll
  for (int c = 0; c < 2; ++c) {
    qf[c].q[0] = *(const u32x4*)(qr + c * 16 + hi * 4);
    qf[c].q[1] = *(const u32x4*)(qr + c * 16 + 8 + hi * 4);
  }

  // Per-thread staging assignments (16B chunks).
  const int krow = tid >> 3, kseg = tid & 7;   // K: 32 rows x 8 chunks of 16B
  const int vrow = tid >> 2, vseg = tid & 3;   // V: 64 rows x 4 chunks of 16B
  const uint32_t kdst = (uint32_t)(size_t)&Ks[krow][kseg * 8];
  const uint32_t vdst = (uint32_t)(size_t)&Vs[vrow][vseg * 8];

  float mrow[8], lrow[8];
  v8f o[4];
#pragma unroll
  for (int i = 0; i < 8; ++i) { mrow[i] = -1e30f; lrow[i] = 0.f; }
#pragma unroll
  for (int nt = 0; nt < 4; ++nt)
#pragma unroll
    for (int i = 0; i < 8; ++i) o[nt][i] = 0.f;

  for (int s0 = 0; s0 < T_; s0 += 32) {
    // ---- async-stage K and V tiles into LDS (bypasses VGPRs, ASYNCcnt) ----
    {
      uint64_t kga = (uint64_t)(size_t)(kbase + (size_t)(s0 + krow) * HD_ + kseg * 8);
      uint64_t vga = (uint64_t)(size_t)(vbase + (size_t)vrow * T_ + s0 + vseg * 8);
      asm volatile("global_load_async_to_lds_b128 %0, %1, off"
                   :: "v"(kdst), "v"(kga) : "memory");
      asm volatile("global_load_async_to_lds_b128 %0, %1, off"
                   :: "v"(vdst), "v"(vga) : "memory");
      asm volatile("s_wait_asynccnt 0x0" ::: "memory");
    }
    __syncthreads();

    // ---- scores: Q @ K^T over HD=64 (two K-chunks), two 16-col subtiles ----
    Frag kf[2][2];
#pragma unroll
    for (int sub = 0; sub < 2; ++sub) {
      const uint32_t* kr = (const uint32_t*)&Ks[sub * 16 + ln][0];
#pragma unroll
      for (int c = 0; c < 2; ++c) {
        kf[sub][c].q[0] = *(const u32x4*)(kr + c * 16 + hi * 4);
        kf[sub][c].q[1] = *(const u32x4*)(kr + c * 16 + 8 + hi * 4);
      }
    }
    v8f z;
#pragma unroll
    for (int i = 0; i < 8; ++i) z[i] = 0.f;
    v8f sA = __builtin_amdgcn_wmma_f32_16x16x32_bf16(false, qf[0].v, false, kf[0][0].v,
                                                     (short)0, z, false, false);
    sA = __builtin_amdgcn_wmma_f32_16x16x32_bf16(false, qf[1].v, false, kf[0][1].v,
                                                 (short)0, sA, false, false);
    v8f sB = __builtin_amdgcn_wmma_f32_16x16x32_bf16(false, qf[0].v, false, kf[1][0].v,
                                                     (short)0, z, false, false);
    sB = __builtin_amdgcn_wmma_f32_16x16x32_bf16(false, qf[1].v, false, kf[1][1].v,
                                                 (short)0, sB, false, false);

    int ok0 = pm[s0 + ln];
    int ok1 = pm[s0 + 16 + ln];
    float sv[2][8];
#pragma unroll
    for (int i = 0; i < 8; ++i) {
      sv[0][i] = ok0 ? sA[i] * 0.125f : -1e30f;     // scale = HD^-0.5
      sv[1][i] = ok1 ? sB[i] * 0.125f : -1e30f;
    }

    // ---- online softmax (lane = column, VGPR = row; reduce over 16 lanes) ----
    float cm[8];
#pragma unroll
    for (int i = 0; i < 8; ++i) cm[i] = fmaxf(sv[0][i], sv[1][i]);
#pragma unroll
    for (int off = 1; off < 16; off <<= 1)
#pragma unroll
      for (int i = 0; i < 8; ++i) cm[i] = fmaxf(cm[i], __shfl_xor(cm[i], off, 32));

    float p0[8], p1[8], rs[8];
#pragma unroll
    for (int i = 0; i < 8; ++i) {
      float mn = fmaxf(mrow[i], cm[i]);
      float al = __expf(mrow[i] - mn);
      mrow[i] = mn;
      lrow[i] *= al;
#pragma unroll
      for (int nt = 0; nt < 4; ++nt) o[nt][i] *= al;
      p0[i] = __expf(sv[0][i] - mn);
      p1[i] = __expf(sv[1][i] - mn);
      rs[i] = p0[i] + p1[i];
    }
#pragma unroll
    for (int off = 1; off < 16; off <<= 1)
#pragma unroll
      for (int i = 0; i < 8; ++i) rs[i] += __shfl_xor(rs[i], off, 32);
#pragma unroll
    for (int i = 0; i < 8; ++i) lrow[i] += rs[i];

    // ---- P (C layout) -> A-fragment layout via this wave's private LDS slab ----
#pragma unroll
    for (int i = 0; i < 8; ++i) {
      int row = hi * 8 + i;
      Ps[wave][row][ln]      = f32_to_bf16(p0[i]);
      Ps[wave][row][16 + ln] = f32_to_bf16(p1[i]);
    }
    asm volatile("s_wait_dscnt 0x0" ::: "memory");  // intra-wave LDS RAW

    Frag pa;
    const uint32_t* pr = (const uint32_t*)&Ps[wave][ln][0];
    pa.q[0] = *(const u32x4*)(pr + hi * 4);
    pa.q[1] = *(const u32x4*)(pr + 8 + hi * 4);

    // ---- P @ V from the shared d-major V tile ----
#pragma unroll
    for (int nt = 0; nt < 4; ++nt) {
      Frag vf;
      const uint32_t* vr = (const uint32_t*)&Vs[nt * 16 + ln][0];
      vf.q[0] = *(const u32x4*)(vr + hi * 4);
      vf.q[1] = *(const u32x4*)(vr + 8 + hi * 4);
      o[nt] = __builtin_amdgcn_wmma_f32_16x16x32_bf16(false, pa.v, false, vf.v,
                                                      (short)0, o[nt], false, false);
    }
    __syncthreads();   // before next iteration overwrites Ks/Vs
  }

  float inv[8];
#pragma unroll
  for (int i = 0; i < 8; ++i) inv[i] = 1.0f / lrow[i];
#pragma unroll
  for (int nt = 0; nt < 4; ++nt)
#pragma unroll
    for (int i = 0; i < 8; ++i) {
      int row = hi * 8 + i;
      int col = h * HD_ + nt * 16 + ln;
      abuf[(size_t)(b * T_ + q0 + row) * C_ + col] = f32_to_bf16(o[nt][i] * inv[i]);
    }
}

// ---------------------------------------------------------------------------
// Output projection: attn(bf16)[4096,512] @ Wo(f32) + bo -> f32 d_out.
// ---------------------------------------------------------------------------
__global__ __launch_bounds__(256)
void out_gemm_kernel(const uint16_t* __restrict__ X, const float* __restrict__ W,
                     const float* __restrict__ bias, float* __restrict__ out) {
  __shared__ __align__(16) uint16_t Xs[64][40];
  __shared__ __align__(16) uint16_t Ws[64][40];

  const int tid  = threadIdx.x;
  const int wave = tid >> 5, lane = tid & 31, hi = lane >> 4, ln = lane & 15;
  const int n0 = blockIdx.x * 64, m0 = blockIdx.y * 64;
  const int mt  = wave & 3;
  const int ntb = (wave >> 2) * 2;

  v8f acc[2];
#pragma unroll
  for (int j = 0; j < 2; ++j)
#pragma unroll
    for (int i = 0; i < 8; ++i) acc[j][i] = 0.f;

  for (int kt = 0; kt < C_; kt += 32) {
#pragma unroll
    for (int i = 0; i < 4; ++i) {
      int idx = tid + i * 256;
      int m = idx >> 4, kd = idx & 15;
      ((uint32_t*)&Xs[m][0])[kd] =
          ((const uint32_t*)(X + (size_t)(m0 + m) * C_ + kt))[kd];
      int n = idx & 63, kw = idx >> 6;
      float w0 = W[(size_t)(kt + kw * 2) * C_ + n0 + n];
      float w1 = W[(size_t)(kt + kw * 2 + 1) * C_ + n0 + n];
      ((uint32_t*)&Ws[n][0])[kw] = pack_bf16x2(w0, w1);
    }
    __syncthreads();

    Frag a, b0, b1;
    const uint32_t* ar = (const uint32_t*)&Xs[mt * 16 + ln][0];
    a.q[0] = *(const u32x4*)(ar + hi * 4);
    a.q[1] = *(const u32x4*)(ar + 8 + hi * 4);
    const uint32_t* br0 = (const uint32_t*)&Ws[ntb * 16 + ln][0];
    b0.q[0] = *(const u32x4*)(br0 + hi * 4);
    b0.q[1] = *(const u32x4*)(br0 + 8 + hi * 4);
    const uint32_t* br1 = (const uint32_t*)&Ws[(ntb + 1) * 16 + ln][0];
    b1.q[0] = *(const u32x4*)(br1 + hi * 4);
    b1.q[1] = *(const u32x4*)(br1 + 8 + hi * 4);

    acc[0] = __builtin_amdgcn_wmma_f32_16x16x32_bf16(false, a.v, false, b0.v,
                                                     (short)0, acc[0], false, false);
    acc[1] = __builtin_amdgcn_wmma_f32_16x16x32_bf16(false, a.v, false, b1.v,
                                                     (short)0, acc[1], false, false);
    __syncthreads();
  }

#pragma unroll
  for (int j = 0; j < 2; ++j) {
    int gn = n0 + (ntb + j) * 16 + ln;
    float bvv = bias[gn];
#pragma unroll
    for (int r = 0; r < 8; ++r) {
      int gm = m0 + mt * 16 + hi * 8 + r;
      out[(size_t)gm * C_ + gn] = acc[j][r] + bvv;
    }
  }
}

// ---------------------------------------------------------------------------
extern "C" void kernel_launch(void* const* d_in, const int* in_sizes, int n_in,
                              void* d_out, int out_size, void* d_ws, size_t ws_size,
                              hipStream_t stream) {
  (void)in_sizes; (void)n_in; (void)out_size; (void)ws_size;
  const float* x   = (const float*)d_in[0];
  const float* enc = (const float*)d_in[1];
  const int*   pm  = (const int*)d_in[2];
  const float* Wq  = (const float*)d_in[3];
  const float* bq  = (const float*)d_in[4];
  const float* Wk  = (const float*)d_in[5];
  const float* bk  = (const float*)d_in[6];
  const float* Wv  = (const float*)d_in[7];
  const float* bv  = (const float*)d_in[8];
  const float* Wo  = (const float*)d_in[9];
  const float* bo  = (const float*)d_in[10];
  float* out = (float*)d_out;

  char* ws = (char*)d_ws;
  const size_t elems = (size_t)B_ * H_ * T_ * HD_;   // 2,097,152
  uint16_t* qb = (uint16_t*)(ws + 0 * elems * 2);
  uint16_t* kb = (uint16_t*)(ws + 1 * elems * 2);
  uint16_t* vb = (uint16_t*)(ws + 2 * elems * 2);
  uint16_t* ab = (uint16_t*)(ws + 3 * elems * 2);

  dim3 gg(C_ / 64, MTOT / 64);   // (8, 64) tiles
  dim3 tb(256);
  proj_gemm_kernel<<<gg, tb, 0, stream>>>(x,   Wq, bq, qb, 0);
  proj_gemm_kernel<<<gg, tb, 0, stream>>>(enc, Wk, bk, kb, 0);
  proj_gemm_kernel<<<gg, tb, 0, stream>>>(enc, Wv, bv, vb, 1);
  attn_kernel<<<dim3(B_ * H_ * (T_ / 128)), tb, 0, stream>>>(qb, kb, vb, pm, ab);
  out_gemm_kernel<<<gg, tb, 0, stream>>>(ab, Wo, bo, out);
}
